// MultiHeadAttention_70497593197096
// MI455X (gfx1250) — matrix operand order repaired and checked by reading, not compiled, not run on hardware
//
#include <hip/hip_runtime.h>

// ---------- types ----------
typedef __attribute__((ext_vector_type(16))) __bf16       v16bf;
typedef __attribute__((ext_vector_type(8)))  float        v8f;
typedef __attribute__((ext_vector_type(4)))  unsigned int u32x4;
typedef __attribute__((__vector_size__(4 * sizeof(int)))) int v4i;

union V16 {               // 16 bf16 assembled from two 16B chunks
  v16bf v;
  u32x4 q[2];
};

__device__ __forceinline__ v8f wmma_bf16(v16bf a, v16bf b, v8f c) {
  // D(f32 16x16) = A(bf16 16x32) * B(bf16 32x16) + C
  return __builtin_amdgcn_wmma_f32_16x16x32_bf16(
      /*neg_a=*/false, a, /*neg_b=*/false, b,
      /*c_mod=*/(short)0, c, /*reuse_a=*/false, /*reuse_b=*/false);
}

// ---------- async global->LDS staging (guarded; fallback = load + ds_store) ----------
#if defined(__HIP_DEVICE_COMPILE__) && defined(__has_builtin)
#if __has_builtin(__builtin_amdgcn_global_load_async_to_lds_b128) && \
    __has_builtin(__builtin_amdgcn_s_wait_asynccnt)
#define HAVE_ASYNC_LDS 1
#endif
#endif
#ifndef HAVE_ASYNC_LDS
#define HAVE_ASYNC_LDS 0
#endif

__device__ __forceinline__ void stage16(__bf16* ldst, const __bf16* gsrc) {
#if HAVE_ASYNC_LDS
  __builtin_amdgcn_global_load_async_to_lds_b128(
      (__attribute__((address_space(1))) v4i*)gsrc,
      (__attribute__((address_space(3))) v4i*)ldst, 0, 0);
#else
  *(u32x4*)ldst = *(const u32x4*)gsrc;
#endif
}
__device__ __forceinline__ void stage_wait() {
#if HAVE_ASYNC_LDS
  __builtin_amdgcn_s_wait_asynccnt(0);
#endif
}

// ---------- constants ----------
#define DMODEL 1024
#define NHEADS 16
#define HDIM   64
#define BATCH  2
#define SEQ    2048
#define ROWS   (BATCH * SEQ)        // 4096
#define KC     128                  // k-chunk staged in LDS
#define NCHUNK (DMODEL / KC)        // 8

// ---------- f32 -> bf16 cast ----------
__global__ void mha_cast_bf16(const float* __restrict__ src,
                              __bf16* __restrict__ dst, int n) {
  int i = blockIdx.x * blockDim.x + threadIdx.x;
  if (i < n) dst[i] = (__bf16)src[i];
}

// ---------- cooperative copy of one 64 x 128 W slab (16KB) into LDS ----------
__device__ __forceinline__ void copy_slab(__bf16* slab, const __bf16* __restrict__ W,
                                          int jbase, int kc, int tid) {
  #pragma unroll
  for (int r = 0; r < 4; ++r) {            // 1024 x 16B chunks, 256 threads
    int idx = tid + r * 256;
    int row = idx >> 4;                    // 0..63  (W row jbase+row)
    int c16 = idx & 15;                    // 16B chunk within 256B row slice
    stage16(slab + row * KC + c16 * 8,
            W + (size_t)(jbase + row) * DMODEL + kc + c16 * 8);
  }
}

// ---------- GEMM: out[4096,1024] = A(bf16) * W^T, W row-major [1024,1024](bf16)
// 256 threads = 8 waves; wave -> 16x64 tile; W k-slab staged in LDS (double buffered),
// A operand software-pipelined one k-step ahead in registers.
__global__ __launch_bounds__(256) void mha_gemm_nt(const __bf16* __restrict__ A,
                                                   const __bf16* __restrict__ W,
                                                   float* __restrict__ out) {
  __shared__ __bf16 wslab[2][64 * KC];     // 2 x 16KB
  const int tid  = threadIdx.x;
  const int lane = tid & 31;
  const int wave = tid >> 5;
  const int n = lane & 15;                 // B/C operand lane column
  const int h = lane >> 4;                 // half-wave select
  const int mbase = blockIdx.y * 128 + wave * 16;
  const int jbase = blockIdx.x * 64;
  const __bf16* arow = A + (size_t)(mbase + n) * DMODEL;   // A lane = output row

  copy_slab(wslab[0], W, jbase, 0, tid);   // prologue: chunk 0
  v8f acc[4] = {};
  V16 a_cur;                               // A 16x32: chunks at K = 16h and 16h+8
  a_cur.q[0] = *(const u32x4*)(arow + 16 * h);
  a_cur.q[1] = *(const u32x4*)(arow + 16 * h + 8);
  stage_wait();
  __syncthreads();

  for (int c = 0; c < NCHUNK; ++c) {
    const int kc = c * KC;
    if (c + 1 < NCHUNK)                    // issue next chunk's copy, overlap with compute
      copy_slab(wslab[(c + 1) & 1], W, jbase, kc + KC, tid);
    const __bf16* slab = wslab[c & 1];
    #pragma unroll
    for (int ks = 0; ks < 4; ++ks) {
      const int knext = (kc + ks * 32 + 32) & (DMODEL - 1);  // wraps (unused) on last step
      V16 a_nxt;                           // pipeline A one step ahead
      a_nxt.q[0] = *(const u32x4*)(arow + knext + 16 * h);
      a_nxt.q[1] = *(const u32x4*)(arow + knext + 16 * h + 8);
      __builtin_prefetch(arow + knext + 256, 0, 1);
      v16bf bv[4];                         // all 4 B tiles from LDS before WMMA group
      #pragma unroll
      for (int t = 0; t < 4; ++t)
        bv[t] = *(const v16bf*)(slab + (16 * t + n) * KC + ks * 32 + 16 * h);
      #pragma unroll
      for (int t = 0; t < 4; ++t)
        acc[t] = wmma_bf16(a_cur.v, bv[t], acc[t]);
      a_cur = a_nxt;
    }
    stage_wait();
    __syncthreads();
  }

  #pragma unroll
  for (int t = 0; t < 4; ++t)
    #pragma unroll
    for (int v = 0; v < 8; ++v)            // C layout: row = v + 8h, col = n
      out[(size_t)(mbase + v + 8 * h) * DMODEL + jbase + 16 * t + n] = acc[t][v];
}

// ---------- RoPE on Q,K (f32) + repack to per-head bf16; V -> transposed bf16 ----------
// Qbf/Kbf: [b*h][s][64]; Vt: [b*h][64][2048]
__global__ void mha_rope_pack(const float* __restrict__ Qf, const float* __restrict__ Kf,
                              const float* __restrict__ Vf, const int* __restrict__ pos,
                              __bf16* __restrict__ Qbf, __bf16* __restrict__ Kbf,
                              __bf16* __restrict__ Vt) {
  int idx = blockIdx.x * blockDim.x + threadIdx.x;   // ROWS * 512 pairs
  if (idx >= ROWS * (DMODEL / 2)) return;
  int row = idx >> 9;            // b*SEQ + s
  int p   = idx & 511;           // h*32 + j
  int hh = p >> 5, j = p & 31;
  int b = row >> 11, s = row & (SEQ - 1);
  int col = hh * HDIM + 2 * j;

  // freqs = THETA^(-j/32), ln(10000)/32 = 0.28782313662
  float ang = (float)pos[s] * __expf(-0.28782313662f * (float)j);
  float sn, cs;
  __sincosf(ang, &sn, &cs);

  size_t src = (size_t)row * DMODEL + col;
  size_t qi  = (((size_t)(b * NHEADS + hh)) * SEQ + s) * HDIM + 2 * j;

  float e = Qf[src], o = Qf[src + 1];
  Qbf[qi]     = (__bf16)(e * cs - o * sn);
  Qbf[qi + 1] = (__bf16)(e * sn + o * cs);

  e = Kf[src]; o = Kf[src + 1];
  Kbf[qi]     = (__bf16)(e * cs - o * sn);
  Kbf[qi + 1] = (__bf16)(e * sn + o * cs);

  size_t vi = (((size_t)(b * NHEADS + hh)) * HDIM + 2 * j) * SEQ + s;
  Vt[vi]       = (__bf16)Vf[src];
  Vt[vi + SEQ] = (__bf16)Vf[src + 1];
}

// ---------- Flash attention: 1 wave per 16-row q tile, k-blocks of 32, online softmax ----------
__global__ __launch_bounds__(128) void mha_attn(const __bf16* __restrict__ Qbf,
                                                const __bf16* __restrict__ Kbf,
                                                const __bf16* __restrict__ Vt,
                                                __bf16* __restrict__ Obf) {
  __shared__ __bf16 lds[4][16 * 32];     // per-wave P transpose staging (1KB each)
  const int lane = threadIdx.x & 31;
  const int wave = threadIdx.x >> 5;
  const int n = lane & 15;
  const int h = lane >> 4;
  const int bh = blockIdx.x;             // 0..31
  const int b  = bh >> 4;
  const int hh = bh & 15;
  const int qbase = (blockIdx.y * 4 + wave) * 16;

  const __bf16* Q = Qbf + (size_t)bh * SEQ * HDIM;
  const __bf16* K = Kbf + (size_t)bh * SEQ * HDIM;
  const __bf16* V = Vt  + (size_t)bh * HDIM * SEQ;
  __bf16* ldsw = lds[wave];

  // Q A-operands for both 32-wide d-steps, kept in registers for whole tile
  V16 qa[2];
  #pragma unroll
  for (int kk = 0; kk < 2; ++kk) {
    const __bf16* qrow = Q + (size_t)(qbase + n) * HDIM + 32 * kk;
    qa[kk].q[0] = *(const u32x4*)(qrow + 16 * h);
    qa[kk].q[1] = *(const u32x4*)(qrow + 16 * h + 8);
  }

  v8f acc[4] = {};
  float mrow[8], lrow[8];
  #pragma unroll
  for (int v = 0; v < 8; ++v) { mrow[v] = -3.0e38f; lrow[v] = 0.0f; }
  const float scale = 0.125f;            // 1/sqrt(64)

  const int kend = qbase + 16;
  for (int kb = 0; kb < kend; kb += 32) {
    // ---- scores: two 16x16 tiles covering k-cols [kb, kb+32)
    v8f s0 = {}, s1 = {};
    const __bf16* krow0 = K + (size_t)(kb + n) * HDIM;
    const __bf16* krow1 = K + (size_t)(kb + 16 + n) * HDIM;
    #pragma unroll
    for (int kk = 0; kk < 2; ++kk) {
      v16bf b0 = *(const v16bf*)(krow0 + 32 * kk + 16 * h);
      v16bf b1 = *(const v16bf*)(krow1 + 32 * kk + 16 * h);
      s0 = wmma_bf16(qa[kk].v, b0, s0);
      s1 = wmma_bf16(qa[kk].v, b1, s1);
    }
    // ---- issue V B-operand loads early so they fly under the softmax VALU work
    v16bf vb[4];
    #pragma unroll
    for (int t = 0; t < 4; ++t)
      vb[t] = *(const v16bf*)(V + (size_t)(16 * t + n) * SEQ + kb + 16 * h);
    // ---- causal mask + online softmax (row reductions across the 16-lane half)
    #pragma unroll
    for (int v = 0; v < 8; ++v) {
      const int qr = qbase + v + 8 * h;
      float a0 = s0[v] * scale;
      float a1 = s1[v] * scale;
      if (kb + n > qr)      a0 = -3.0e38f;
      if (kb + 16 + n > qr) a1 = -3.0e38f;
      float rm = fmaxf(a0, a1);
      rm = fmaxf(rm, __shfl_xor(rm, 1, 32));
      rm = fmaxf(rm, __shfl_xor(rm, 2, 32));
      rm = fmaxf(rm, __shfl_xor(rm, 4, 32));
      rm = fmaxf(rm, __shfl_xor(rm, 8, 32));
      const float newm = fmaxf(mrow[v], rm);
      const float p0 = __expf(a0 - newm);
      const float p1 = __expf(a1 - newm);
      float rs = p0 + p1;
      rs += __shfl_xor(rs, 1, 32);
      rs += __shfl_xor(rs, 2, 32);
      rs += __shfl_xor(rs, 4, 32);
      rs += __shfl_xor(rs, 8, 32);
      const float alpha = __expf(mrow[v] - newm);
      lrow[v] = lrow[v] * alpha + rs;
      mrow[v] = newm;
      #pragma unroll
      for (int t = 0; t < 4; ++t) acc[t][v] *= alpha;   // rescale rows v+8h
      // stash P (C-layout) into LDS for transpose to A-layout
      ldsw[(v + 8 * h) * 32 + n]      = (__bf16)p0;
      ldsw[(v + 8 * h) * 32 + 16 + n] = (__bf16)p1;
    }
    asm volatile("s_wait_dscnt 0x0" ::: "memory");
    // ---- P as A-operand (16x32 bf16), read back in A layout
    V16 pa;
    pa.q[0] = *(const u32x4*)(ldsw + n * 32 + 16 * h);
    pa.q[1] = *(const u32x4*)(ldsw + n * 32 + 16 * h + 8);
    // ---- O += P * V
    #pragma unroll
    for (int t = 0; t < 4; ++t)
      acc[t] = wmma_bf16(pa.v, vb[t], acc[t]);
  }

  // ---- epilogue: normalize and write bf16 [row = b*SEQ+s][h*64+d]
  #pragma unroll
  for (int v = 0; v < 8; ++v) {
    const float inv = 1.0f / lrow[v];
    const size_t row = (size_t)(b * SEQ + qbase + v + 8 * h);
    #pragma unroll
    for (int t = 0; t < 4; ++t)
      Obf[row * DMODEL + hh * HDIM + 16 * t + n] = (__bf16)(acc[t][v] * inv);
  }
}

// ---------- launcher ----------
extern "C" void kernel_launch(void* const* d_in, const int* in_sizes, int n_in,
                              void* d_out, int out_size, void* d_ws, size_t ws_size,
                              hipStream_t stream) {
  const float* x  = (const float*)d_in[0];
  const float* Wq = (const float*)d_in[1];
  const float* Wk = (const float*)d_in[2];
  const float* Wv = (const float*)d_in[3];
  const float* Wo = (const float*)d_in[4];
  const int* pos  = (const int*)d_in[6];     // token_positions; mask handled analytically
  float* out = (float*)d_out;

  char* ws = (char*)d_ws;
  const size_t MB = (size_t)1 << 20;
  __bf16* Xbf = (__bf16*)(ws);               //  8 MB  [4096][1024]
  __bf16* Wqb = (__bf16*)(ws + 8 * MB);      //  2 MB each
  __bf16* Wkb = (__bf16*)(ws + 10 * MB);
  __bf16* Wvb = (__bf16*)(ws + 12 * MB);
  __bf16* Wob = (__bf16*)(ws + 14 * MB);
  float*  Qf  = (float*)(ws + 16 * MB);      // 16 MB each
  float*  Kf  = (float*)(ws + 32 * MB);
  float*  Vf  = (float*)(ws + 48 * MB);
  __bf16* Qbf = (__bf16*)(ws + 64 * MB);     //  8 MB each
  __bf16* Kbf = (__bf16*)(ws + 72 * MB);
  __bf16* Vt  = (__bf16*)(ws + 80 * MB);
  __bf16* Obf = (__bf16*)(ws + 88 * MB);     // total 96 MB

  const int NX = ROWS * DMODEL;              // 4,194,304
  const int NW = DMODEL * DMODEL;            // 1,048,576
  mha_cast_bf16<<<(NX + 255) / 256, 256, 0, stream>>>(x,  Xbf, NX);
  mha_cast_bf16<<<(NW + 255) / 256, 256, 0, stream>>>(Wq, Wqb, NW);
  mha_cast_bf16<<<(NW + 255) / 256, 256, 0, stream>>>(Wk, Wkb, NW);
  mha_cast_bf16<<<(NW + 255) / 256, 256, 0, stream>>>(Wv, Wvb, NW);
  mha_cast_bf16<<<(NW + 255) / 256, 256, 0, stream>>>(Wo, Wob, NW);

  dim3 gGrid(DMODEL / 64, ROWS / 128);       // (16, 32)
  mha_gemm_nt<<<gGrid, 256, 0, stream>>>(Xbf, Wqb, Qf);
  mha_gemm_nt<<<gGrid, 256, 0, stream>>>(Xbf, Wkb, Kf);
  mha_gemm_nt<<<gGrid, 256, 0, stream>>>(Xbf, Wvb, Vf);

  const int NR = ROWS * (DMODEL / 2);        // 2,097,152
  mha_rope_pack<<<(NR + 255) / 256, 256, 0, stream>>>(Qf, Kf, Vf, pos, Qbf, Kbf, Vt);

  mha_attn<<<dim3(BATCH * NHEADS, SEQ / 64), 128, 0, stream>>>(Qbf, Kbf, Vt, Obf);

  mha_gemm_nt<<<gGrid, 256, 0, stream>>>(Obf, Wob, out);
}
